// TGNN_69355131895969
// MI455X (gfx1250) — compile-verified, matrix operand order checked by reading, not run on hardware
//
#include <hip/hip_runtime.h>
#include <math.h>

typedef __attribute__((ext_vector_type(16))) _Float16 v16h;
typedef __attribute__((ext_vector_type(8)))  float    v8f;
typedef __attribute__((ext_vector_type(2)))  float    v2f;

#define SEQ      65536
#define FEAT     512
#define THRES1   0.8f
#define THRES_UP 0.5f
#define MAXF     60

#if defined(__has_builtin)
#if __has_builtin(__builtin_amdgcn_wmma_f32_16x16x4_f32)
#define HAVE_WMMA_F32X4 1
#endif
#endif

// ---------------------------------------------------------------------------
// Kernel 1: a[i] = sigmoid(dot(h[i,:], W) + b) via WMMA.
// One wave (32 lanes) computes 16 rows. B fragment broadcasts W[k] across all
// 16 N columns, so D[m][n] == dot(row m) for every n; lanes 0 and 16 hold all
// 16 results in their 8 accumulator VGPRs.
//
// Preferred path: native f32 V_WMMA_F32_16X16X4_F32 (no f32->f16 cvt in the
// loop, exact f32 precision).
//   A 16x4 f32 layout (ISA 7.12.2): lane l (g=l>>4): V0=K(2g), V1=K(2g+1).
//   B 4x16 f32 (C/D half-split pattern): lane l: V0=row(2g), V1=row(2g+1),
//   N = l&15 -> same v2f index as A, loaded from W.
// Fallback: codegen-confirmed f16 16x16x32 path with on-the-fly conversion.
// ---------------------------------------------------------------------------
__global__ __launch_bounds__(256) void gemv_wmma_sigmoid(
    const float* __restrict__ h, const float* __restrict__ W,
    const float* __restrict__ bias, float* __restrict__ aout) {
  const int lane = threadIdx.x & 31;
  const int wave = (blockIdx.x * blockDim.x + threadIdx.x) >> 5;
  const int row0 = wave * 16;
  const int g    = lane >> 4;
  const float* hrow = h + (size_t)(row0 + (lane & 15)) * FEAT;

  v8f c = {0.f, 0.f, 0.f, 0.f, 0.f, 0.f, 0.f, 0.f};

#if HAVE_WMMA_F32X4
  const float* ap = hrow + 2 * g;   // K = kb+2g, kb+2g+1
  const float* wp = W + 2 * g;      // broadcast-B: same K indexing
#pragma unroll 8
  for (int kb = 0; kb < FEAT; kb += 4) {
    v2f A = *(const v2f*)(ap + kb);
    v2f B = *(const v2f*)(wp + kb);
    c = __builtin_amdgcn_wmma_f32_16x16x4_f32(
        /*neg_a=*/false, A, /*neg_b=*/false, B,
        /*c_mod=*/(short)0, c, /*reuse_a=*/false, /*reuse_b=*/false);
  }
#else
  for (int kb = 0; kb < FEAT; kb += 32) {
    const int base1 = kb + 8 * g;        // K = kb+8g    .. +7 -> elems 0..7
    const int base2 = kb + 16 + 8 * g;   // K = kb+16+8g .. +7 -> elems 8..15
    const float* wp = W + kb + 16 * g;   // K = kb+16g .. +15  -> B elems 0..15
    v16h A, B;
#pragma unroll
    for (int e = 0; e < 8; ++e)  A[e]     = (_Float16)hrow[base1 + e];
#pragma unroll
    for (int e = 0; e < 8; ++e)  A[8 + e] = (_Float16)hrow[base2 + e];
#pragma unroll
    for (int e = 0; e < 16; ++e) B[e]     = (_Float16)wp[e];
    c = __builtin_amdgcn_wmma_f32_16x16x32_f16(
        false, A, false, B, (short)0, c, false, false);
  }
#endif

  const float bs = bias[0];
  if ((lane & 15) == 0) {
    // lane 0: c[r] = dot(row0+r); lane 16: c[r] = dot(row0+8+r)
    const int rbase = row0 + 8 * g;
#pragma unroll
    for (int r = 0; r < 8; ++r) {
      float v = c[r] + bs;
      aout[rbase + r] = 1.0f / (1.0f + __expf(-v));
    }
  }
}

// ---------------------------------------------------------------------------
// Kernel 2: elementwise pass 1.
//   gate_i = (u_pred_i >= 0.5);  up_hat_i = gate ? runmax(u_pred[s..i]) : u_pred_i
//   (up_hat_i >= 0.5 <=> u_pred_i >= 0.5, so the gate is elementwise).
//   Emit u_pred passthrough + up_hats, and alpha-recurrence coefficients
//   A = up_hat, B = (1-up_hat)*a. Zero loss/cnt slots.
// ---------------------------------------------------------------------------
__global__ __launch_bounds__(256) void pass_uphat(
    const float* __restrict__ up, const float* __restrict__ a,
    float* __restrict__ out, float* __restrict__ Aarr, float* __restrict__ Barr) {
  const int i = blockIdx.x * 256 + threadIdx.x;
  const float u = up[i];
  float uh;
  if (u >= THRES_UP) {
    float m = u;
    int j = i;
    while (j > 0 && up[j - 1] >= THRES_UP) { --j; m = fmaxf(m, up[j]); }
    uh = m;                                  // segmented running max
  } else {
    uh = u;
  }
  out[2 * SEQ + i] = u;    // u_pred passthrough (output 3)
  out[3 * SEQ + i] = uh;   // up_hats           (output 4)
  Aarr[i] = uh;
  Barr[i] = (1.0f - uh) * a[i];
  if (i == 0) { out[4 * SEQ] = 0.0f; out[4 * SEQ + 1] = 0.0f; }
}

// ---------------------------------------------------------------------------
// Affine scan machinery: x_i = A_i * x_{i-1} + B_i, x_{-1} = 0.
// Block-local inclusive scan of affine maps (Hillis-Steele in LDS), then a
// serial 256-entry carry combine, then apply.
// ---------------------------------------------------------------------------
__global__ __launch_bounds__(256) void affine_scan_block(
    float* __restrict__ Aarr, float* __restrict__ Barr,
    float* __restrict__ blkP, float* __restrict__ blkS) {
  __shared__ float sA[256], sB[256];
  const int t = threadIdx.x;
  const int i = blockIdx.x * 256 + t;
  sA[t] = Aarr[i];
  sB[t] = Barr[i];
  __syncthreads();
#pragma unroll
  for (int off = 1; off < 256; off <<= 1) {
    float a1 = 0.f, b1 = 0.f;
    const bool has = (t >= off);
    if (has) { a1 = sA[t - off]; b1 = sB[t - off]; }
    const float a2 = sA[t], b2 = sB[t];
    __syncthreads();
    if (has) { sA[t] = a2 * a1; sB[t] = a2 * b1 + b2; }   // later ∘ earlier
    __syncthreads();
  }
  Aarr[i] = sA[t];   // in-place: (P_i, S_i) composite from block start
  Barr[i] = sB[t];
  if (t == 255) { blkP[blockIdx.x] = sA[255]; blkS[blockIdx.x] = sB[255]; }
}

__global__ void carry_combine(const float* __restrict__ blkP,
                              const float* __restrict__ blkS,
                              float* __restrict__ carry) {
  if (threadIdx.x == 0 && blockIdx.x == 0) {
    float c = 0.0f;                       // x_{-1} = 0
    for (int b = 0; b < 256; ++b) { carry[b] = c; c = blkP[b] * c + blkS[b]; }
  }
}

__global__ __launch_bounds__(256) void affine_apply(
    const float* __restrict__ P, const float* __restrict__ S,
    const float* __restrict__ carry, float* __restrict__ dst) {
  const int i = blockIdx.x * 256 + threadIdx.x;
  dst[i] = P[i] * carry[blockIdx.x] + S[i];
}

// ---------------------------------------------------------------------------
// Kernel: coefficients for the y recurrence.
//   gate open:  y_i = (1-alpha_i) y_{i-1} + alpha_i*up_hat_i
//   gate closed: y_i = 0   (absorbing affine element (0,0))
// ---------------------------------------------------------------------------
__global__ __launch_bounds__(256) void pass_ycoef(
    const float* __restrict__ up, const float* __restrict__ out,
    float* __restrict__ Aarr, float* __restrict__ Barr) {
  const int i = blockIdx.x * 256 + threadIdx.x;
  const float alpha = out[SEQ + i];       // alphas
  const float uh    = out[3 * SEQ + i];   // up_hats
  const bool gate   = up[i] >= THRES_UP;
  Aarr[i] = gate ? (1.0f - alpha) : 0.0f;
  Barr[i] = gate ? (alpha * uh)   : 0.0f;
}

// ---------------------------------------------------------------------------
// Kernel: loss/cnt at falling edges. At fall f, the pending l_c is the most
// recent label>=0.8 assignment at j in (prev_fall, f]; assignment value is
// -1 if gate closed at j, else (y_j - label_j)^2. Flush per reference rules.
// ---------------------------------------------------------------------------
__global__ __launch_bounds__(256) void pass_falls(
    const float* __restrict__ up, const float* __restrict__ label,
    const float* __restrict__ ys, const float* __restrict__ thres2,
    float* __restrict__ out) {
  const int i = blockIdx.x * 256 + threadIdx.x;
  if (i == 0) return;
  if (!(up[i] < THRES_UP && up[i - 1] >= THRES_UP)) return;   // fall at i

  bool found = false;
  int jf = -1;
  int j = i;
  while (true) {
    if (label[j] >= THRES1) { found = true; jf = j; break; }
    if (j == 0) break;
    --j;
    // stop (exclusive) at the previous fall: its assignment was reset away
    const bool jfall = (up[j] < THRES_UP) && (j > 0) && (up[j - 1] >= THRES_UP);
    if (jfall) break;
  }

  bool is_neg = false;
  float lc = 0.0f;
  if (found) {
    if (up[jf] < THRES_UP) {              // up_hat_j < 0.5  ->  l_c = -1
      is_neg = true;
    } else {
      const float d = ys[jf] - label[jf];
      lc = d * d;
    }
  }

  float contrib = 0.0f;
  int add = 0;
  if (!is_neg) {
    if (found && lc != 0.0f) {            // add_c path (R = 1.0)
      contrib = lc; add = 1;
    } else {                              // add_e path
      const float ypre = ys[i - 1];
      if (ypre >= THRES1) {
        const float t2 = thres2[i < (MAXF - 1) ? i : (MAXF - 1)];
        const float d = ypre - t2;
        contrib = d * d; add = 1;
      }
    }
  }
  if (add) {
    atomicAdd(&out[4 * SEQ], contrib);
    atomicAdd(&out[4 * SEQ + 1], 1.0f);
  }
}

// ---------------------------------------------------------------------------
// d_out layout (floats): ys[SEQ] | alphas[SEQ] | u_pred[SEQ] | up_hats[SEQ]
//                        | loss | cnt
// ws layout (floats):    a[SEQ] | A[SEQ] | B[SEQ] | blkP[256] | blkS[256]
//                        | carry[256]
// ---------------------------------------------------------------------------
extern "C" void kernel_launch(void* const* d_in, const int* in_sizes, int n_in,
                              void* d_out, int out_size, void* d_ws, size_t ws_size,
                              hipStream_t stream) {
  (void)in_sizes; (void)n_in; (void)out_size; (void)ws_size;
  const float* h      = (const float*)d_in[0];
  const float* W      = (const float*)d_in[1];
  const float* b      = (const float*)d_in[2];
  /* d_in[3] = u (unused by the reference loss) */
  const float* u_pred = (const float*)d_in[4];
  const float* label  = (const float*)d_in[5];
  const float* thres2 = (const float*)d_in[6];

  float* out   = (float*)d_out;
  float* ws    = (float*)d_ws;
  float* a     = ws;                 // SEQ
  float* Aarr  = ws + SEQ;           // SEQ
  float* Barr  = ws + 2 * SEQ;       // SEQ
  float* blkP  = ws + 3 * SEQ;       // 256
  float* blkS  = blkP + 256;         // 256
  float* carry = blkS + 256;         // 256

  // 1) memory-bound WMMA GEMV + sigmoid: 4096 waves, 8 waves/block
  gemv_wmma_sigmoid<<<SEQ / 16 / 8, 256, 0, stream>>>(h, W, b, a);

  // 2) gate / up_hat / alpha coefficients (+ zero loss & cnt)
  pass_uphat<<<256, 256, 0, stream>>>(u_pred, a, out, Aarr, Barr);

  // 3) alpha affine scan -> alphas at out+SEQ
  affine_scan_block<<<256, 256, 0, stream>>>(Aarr, Barr, blkP, blkS);
  carry_combine<<<1, 1, 0, stream>>>(blkP, blkS, carry);
  affine_apply<<<256, 256, 0, stream>>>(Aarr, Barr, carry, out + SEQ);

  // 4) y segmented affine scan -> ys at out+0
  pass_ycoef<<<256, 256, 0, stream>>>(u_pred, out, Aarr, Barr);
  affine_scan_block<<<256, 256, 0, stream>>>(Aarr, Barr, blkP, blkS);
  carry_combine<<<1, 1, 0, stream>>>(blkP, blkS, carry);
  affine_apply<<<256, 256, 0, stream>>>(Aarr, Barr, carry, out);

  // 5) loss / cnt at falling edges
  pass_falls<<<256, 256, 0, stream>>>(u_pred, label, out, thres2, out);
}
